// ChangeDecoder_88373247083079
// MI455X (gfx1250) — compile-verified, matrix operand order checked by reading, not run on hardware
//
#include <hip/hip_runtime.h>
#include <hip/hip_bf16.h>
#include <math.h>

// ---------------------------------------------------------------------------
// CDNA5 (gfx1250) wave32 WMMA decoder implementation.
// All GEMM-shaped work lowers to v_wmma_f32_16x16x32_bf16 (fp32 accumulate).
// ---------------------------------------------------------------------------

typedef __attribute__((ext_vector_type(16))) __bf16 bf16x16;
typedef __attribute__((ext_vector_type(8)))  float  f32x8;

#define DEVFN static __device__ __forceinline__

DEVFN float dsigmoid(float x){ return 1.f/(1.f+__expf(-x)); }
DEVFN float dsilu(float x){ return x*dsigmoid(x); }
DEVFN float dsoftplus(float x){ return (x>20.f)?x:log1pf(__expf(x)); }
DEVFN float dact(float x,int act){
  if(act==1) return fmaxf(x,0.f);
  if(act==2) return dsilu(x);
  if(act==3) return dsigmoid(x);
  if(act==4) return dsoftplus(x);
  return x;
}
DEVFN int iclampi(int v,int lo,int hi){ return v<lo?lo:(v>hi?hi:v); }
DEVFN float wred_sum(float v){ for(int o=16;o;o>>=1) v+=__shfl_xor(v,o,32); return v; }
DEVFN float wred_max(float v){ for(int o=16;o;o>>=1) v=fmaxf(v,__shfl_xor(v,o,32)); return v; }

// ---------------------------------------------------------------------------
// Generic WMMA GEMM:  O[bz] = act( (alpha*(A op W^T) + bias) [*bnScale+bnShift] )
//   A: (M,K) row-major, row stride lda   (AT==1 -> A stored transposed: A[k*lda+m])
//   W: WT==0 -> W[n*ldw+k] (PyTorch Linear weight (N,K));  WT==1 -> W[k*ldw+n]
//   batch bz along grid.z; weights/bias advance every `bdiv` batches.
// One wave per 16x(16*NT) output strip, K tiled by 32 (bf16 WMMA, fp32 acc).
// Interior tiles: unguarded float2 fragment loads (branch-free hot loop).
// Edge tiles / K tail: clamped-index loads with multiply-by-mask (no branches).
// ---------------------------------------------------------------------------
template<int WT, int AT, int NT>
__global__ void k_gemm(const float* __restrict__ A, const float* __restrict__ W,
                       float* __restrict__ O,
                       const float* __restrict__ bias, const float* __restrict__ scale,
                       const float* __restrict__ shift,
                       int M, int N, int K, int lda, int ldw, int ldo,
                       long long sA, long long sW, long long sO, long long sBias,
                       float alpha, float scale_mul, int bdiv, int act)
{
  const int mt  = blockIdx.x * 16;
  const int nt0 = blockIdx.y * 16 * NT;
  const int bz  = blockIdx.z;
  const float* Ab = A + (long long)bz * sA;
  const float* Wb = W + (long long)(bz / bdiv) * sW;
  float*       Ob = O + (long long)bz * sO;
  const float* Bb = bias ? bias + (long long)(bz / bdiv) * sBias : nullptr;

  const int lane = threadIdx.x & 31;
  const int half = lane >> 4;   // which K-half of the 32-wide tile this lane feeds
  const int lm   = lane & 15;
  const int am   = mt + lm;     // A row fed by this lane

  // ISA 16-bit A-matrix 16x32 layout: vgpr v<4 -> K=2v(+8*half), v>=4 -> K=16+2(v-4)(+8*half)
  int koff[8];
#pragma unroll
  for (int v = 0; v < 8; ++v)
    koff[v] = ((v < 4) ? (2 * v) : (16 + 2 * (v - 4))) + 8 * half;

  f32x8 acc[NT];
#pragma unroll
  for (int t = 0; t < NT; ++t) acc[t] = (f32x8){};

  const bool interior = (mt + 16 <= M) && (nt0 + 16 * NT <= N);
  int k0 = 0;

  if (interior) {
    for (; k0 + 32 <= K; k0 += 32) {
      bf16x16 af;
      if (AT == 0) {
        const float* ap = Ab + (long long)am * lda + k0;
#pragma unroll
        for (int v = 0; v < 8; ++v) {
          const float2 p = *(const float2*)(ap + koff[v]);
          af[2 * v] = (__bf16)p.x; af[2 * v + 1] = (__bf16)p.y;
        }
      } else {
#pragma unroll
        for (int v = 0; v < 8; ++v) {
          const int k = k0 + koff[v];
          af[2 * v]     = (__bf16)Ab[(long long)k * lda + am];
          af[2 * v + 1] = (__bf16)Ab[(long long)(k + 1) * lda + am];
        }
      }
#pragma unroll
      for (int t = 0; t < NT; ++t) {
        const int bn = nt0 + 16 * t + lm;
        bf16x16 bf;
        if (WT == 0) {
          const float* wp = Wb + (long long)bn * ldw + k0;
#pragma unroll
          for (int v = 0; v < 8; ++v) {
            const float2 p = *(const float2*)(wp + koff[v]);
            bf[2 * v] = (__bf16)p.x; bf[2 * v + 1] = (__bf16)p.y;
          }
        } else {
#pragma unroll
          for (int v = 0; v < 8; ++v) {
            const int k = k0 + koff[v];
            bf[2 * v]     = (__bf16)Wb[(long long)k * ldw + bn];
            bf[2 * v + 1] = (__bf16)Wb[(long long)(k + 1) * ldw + bn];
          }
        }
        acc[t] = __builtin_amdgcn_wmma_f32_16x16x32_bf16(false, af, false, bf,
                                                         (short)0, acc[t], false, false);
      }
    }
  }

  // masked tail / edge tiles: clamped indices + select masks, no divergent loads
  for (; k0 < K; k0 += 32) {
    bf16x16 af;
    const int amc = am < M ? am : M - 1;
    const float ma = am < M ? 1.f : 0.f;
#pragma unroll
    for (int v = 0; v < 8; ++v) {
      const int k = k0 + koff[v];
      const int kc0 = k < K ? k : K - 1;
      const int kc1 = (k + 1) < K ? (k + 1) : K - 1;
      const float m0 = (k < K)     ? ma : 0.f;
      const float m1 = ((k + 1) < K) ? ma : 0.f;
      float a0, a1;
      if (AT == 0) { a0 = Ab[(long long)amc * lda + kc0]; a1 = Ab[(long long)amc * lda + kc1]; }
      else         { a0 = Ab[(long long)kc0 * lda + amc]; a1 = Ab[(long long)kc1 * lda + amc]; }
      af[2 * v] = (__bf16)(a0 * m0); af[2 * v + 1] = (__bf16)(a1 * m1);
    }
#pragma unroll
    for (int t = 0; t < NT; ++t) {
      const int bn = nt0 + 16 * t + lm;
      const int bnc = bn < N ? bn : N - 1;
      const float mb = bn < N ? 1.f : 0.f;
      bf16x16 bf;
#pragma unroll
      for (int v = 0; v < 8; ++v) {
        const int k = k0 + koff[v];
        const int kc0 = k < K ? k : K - 1;
        const int kc1 = (k + 1) < K ? (k + 1) : K - 1;
        const float m0 = (k < K)       ? mb : 0.f;
        const float m1 = ((k + 1) < K) ? mb : 0.f;
        float b0, b1;
        if (WT == 0) { b0 = Wb[(long long)bnc * ldw + kc0]; b1 = Wb[(long long)bnc * ldw + kc1]; }
        else         { b0 = Wb[(long long)kc0 * ldw + bnc]; b1 = Wb[(long long)kc1 * ldw + bnc]; }
        bf[2 * v] = (__bf16)(b0 * m0); bf[2 * v + 1] = (__bf16)(b1 * m1);
      }
      acc[t] = __builtin_amdgcn_wmma_f32_16x16x32_bf16(false, af, false, bf,
                                                       (short)0, acc[t], false, false);
    }
  }

  // epilogue: D element v -> (m = mt + v + 8*half, n = nt0 + 16*t + lm)
#pragma unroll
  for (int t = 0; t < NT; ++t) {
    const int n = nt0 + 16 * t + lm;
    if (n < N) {
      const float bv = Bb ? Bb[n] : 0.f;
      const float sc = scale ? scale[n] * scale_mul : 1.f;
      const float sh = shift ? shift[n] : 0.f;
#pragma unroll
      for (int v = 0; v < 8; ++v) {
        const int m = mt + v + 8 * half;
        if (m < M) {
          float x = acc[t][v] * alpha + bv;
          if (scale) x = x * sc + sh;
          Ob[(long long)m * ldo + n] = dact(x, act);
        }
      }
    }
  }
}

// ---------------------------------------------------------------------------
// Layout / elementwise kernels
// ---------------------------------------------------------------------------
__global__ void k_cat_nchw2nhwc(const float* __restrict__ x1, const float* __restrict__ x2,
                                float* __restrict__ o, int B, int C1, int C2, int HW)
{
  long long i = (long long)blockIdx.x * blockDim.x + threadIdx.x;
  const int C = C1 + C2;
  long long total = (long long)B * HW * C;
  if (i >= total) return;
  int c = (int)(i % C); long long r = i / C;
  int p = (int)(r % HW); int b = (int)(r / HW);
  float v = (c < C1) ? x1[((long long)b * C1 + c) * HW + p]
                     : x2[((long long)b * C2 + (c - C1)) * HW + p];
  o[i] = v;
}

__global__ void k_cat_nhwc(const float* __restrict__ a, const float* __restrict__ b2,
                           float* __restrict__ o, long long rows, int C1, int C2)
{
  long long i = (long long)blockIdx.x * blockDim.x + threadIdx.x;
  const int C = C1 + C2;
  long long total = rows * C;
  if (i >= total) return;
  int c = (int)(i % C); long long r = i / C;
  o[i] = (c < C1) ? a[r * C1 + c] : b2[r * C2 + (c - C1)];
}

__global__ void k_nhwc2nchw(const float* __restrict__ x, float* __restrict__ o,
                            int B, int C, int HW)
{
  long long i = (long long)blockIdx.x * blockDim.x + threadIdx.x;
  long long total = (long long)B * C * HW;
  if (i >= total) return;
  int p = (int)(i % HW); long long r = i / HW;
  int c = (int)(r % C); int b = (int)(r / C);
  o[i] = x[((long long)b * HW + p) * C + c];
}

__global__ void k_add_inplace(float* __restrict__ o, const float* __restrict__ a, long long n)
{
  long long i = (long long)blockIdx.x * blockDim.x + threadIdx.x;
  if (i < n) o[i] += a[i];
}

__global__ void k_layernorm(const float* __restrict__ x, float* __restrict__ o,
                            const float* __restrict__ g, const float* __restrict__ b,
                            int rows, int C)
{
  const int row = blockIdx.x;
  const int lane = threadIdx.x;
  if (row >= rows) return;
  const float* xr = x + (long long)row * C;
  float s = 0.f, s2 = 0.f;
  for (int c = lane; c < C; c += 32) { float v = xr[c]; s += v; s2 += v * v; }
  s = wred_sum(s); s2 = wred_sum(s2);
  const float mean = s / C;
  const float var  = s2 / C - mean * mean;
  const float r    = rsqrtf(var + 1e-5f);
  float* orow = o + (long long)row * C;
  for (int c = lane; c < C; c += 32) orow[c] = (xr[c] - mean) * r * g[c] + b[c];
}

// depthwise 3x3 SAME conv on NHWC (input channel stride ldx), fused bias/BN/act
__global__ void k_dwconv3x3(const float* __restrict__ x, const float* __restrict__ w,
                            const float* __restrict__ bias, const float* __restrict__ scale,
                            const float* __restrict__ shift, float scale_mul,
                            float* __restrict__ o, int B, int H, int W, int C,
                            int ldx, int act)
{
  long long i = (long long)blockIdx.x * blockDim.x + threadIdx.x;
  long long total = (long long)B * H * W * C;
  if (i >= total) return;
  int c = (int)(i % C); long long r = i / C;
  int wx = (int)(r % W); long long q = r / W;
  int hy = (int)(q % H); int b = (int)(q / H);
  float acc = bias ? bias[c] : 0.f;
#pragma unroll
  for (int ky = 0; ky < 3; ++ky) {
    int yy = hy + ky - 1;
    if (yy < 0 || yy >= H) continue;
#pragma unroll
    for (int kx = 0; kx < 3; ++kx) {
      int xx = wx + kx - 1;
      if (xx < 0 || xx >= W) continue;
      acc += x[(((long long)b * H + yy) * W + xx) * ldx + c] * w[c * 9 + ky * 3 + kx];
    }
  }
  if (scale) acc = acc * scale[c] * scale_mul + shift[c];
  o[i] = dact(acc, act);
}

// xc NHWC (B,H,W,Dh) -> xs (4,B,Dh,L) four scan directions
__global__ void k_build_xs(const float* __restrict__ xc, float* __restrict__ xs,
                           int B, int H, int W, int Dh)
{
  long long i = (long long)blockIdx.x * blockDim.x + threadIdx.x;
  long long total = (long long)B * H * W * Dh;
  if (i >= total) return;
  int d = (int)(i % Dh); long long r = i / Dh;
  int w = (int)(r % W); long long q = r / W;
  int h = (int)(q % H); int b = (int)(q / H);
  const float v = xc[i];
  const int L = H * W, lhw = h * W + w, lwh = w * H + h;
  const long long sB = (long long)Dh * L, sK = (long long)B * sB;
  const long long base = (long long)b * sB + (long long)d * L;
  xs[0 * sK + base + lhw]          = v;
  xs[1 * sK + base + lwh]          = v;
  xs[2 * sK + base + (L - 1 - lhw)] = v;
  xs[3 * sK + base + (L - 1 - lwh)] = v;
}

// sequential selective scan; one thread per (k,b,d), 16-wide state in registers
__global__ void k_scan(const float* __restrict__ xs,   // (4,B,Dh,L)  u
                       const float* __restrict__ dtT,  // (4,B,L,Dh)
                       const float* __restrict__ xdbl, // (4,B,L,40)  B at col 8, C at col 24
                       const float* __restrict__ A_log,// (4,Dh,16)
                       const float* __restrict__ Ds,   // (4,Dh)
                       float* __restrict__ ys,         // (4,B,Dh,L)
                       int B, int Dh, int L)
{
  int i = blockIdx.x * blockDim.x + threadIdx.x;
  int total = 4 * B * Dh;
  if (i >= total) return;
  int d = i % Dh; int t = i / Dh; int b = t % B; int k = t / B;

  float Av[16], h[16];
#pragma unroll
  for (int n = 0; n < 16; ++n) { Av[n] = -__expf(A_log[((long long)k * Dh + d) * 16 + n]); h[n] = 0.f; }
  const float Dd = Ds[(long long)k * Dh + d];

  const float* u   = xs  + (((long long)k * B + b) * Dh + d) * L;
  const float* dtp = dtT + ((long long)k * B + b) * L * Dh + d;
  const float* xd  = xdbl + ((long long)k * B + b) * L * 40;
  float*       y   = ys  + (((long long)k * B + b) * Dh + d) * L;

  for (int l = 0; l < L; ++l) {
    __builtin_prefetch(xd + (long long)(l + 16) * 40, 0, 1);
    const float dt = dtp[(long long)l * Dh];
    const float ul = u[l];
    const float du = dt * ul;
    const float* Bp = xd + (long long)l * 40 + 8;
    const float* Cp = Bp + 16;
    float acc = 0.f;
#pragma unroll
    for (int n = 0; n < 16; ++n) {
      h[n] = h[n] * __expf(dt * Av[n]) + du * Bp[n];
      acc += h[n] * Cp[n];
    }
    y[l] = acc + Dd * ul;
  }
}

// combine 4 scan directions -> yt (B, L, Dh) row-major (NHWC over l)
__global__ void k_combine_y(const float* __restrict__ ys, float* __restrict__ yt,
                            int B, int H, int W, int Dh)
{
  long long i = (long long)blockIdx.x * blockDim.x + threadIdx.x;
  long long total = (long long)B * H * W * Dh;
  if (i >= total) return;
  int d = (int)(i % Dh); long long r = i / Dh;
  int w = (int)(r % W); long long q = r / W;
  int h = (int)(q % H); int b = (int)(q / H);
  const int L = H * W, lhw = h * W + w, lwh = w * H + h;
  const long long sB = (long long)Dh * L, sK = (long long)B * sB;
  const long long base = (long long)b * sB + (long long)d * L;
  yt[i] = ys[0 * sK + base + lhw] + ys[2 * sK + base + (L - 1 - lhw)]
        + ys[1 * sK + base + lwh] + ys[3 * sK + base + (L - 1 - lwh)];
}

// y (rows,Dh) *= silu(z) where z = xz[:, Dh:2Dh]
__global__ void k_gate(float* __restrict__ y, const float* __restrict__ xz,
                       long long rows, int Dh)
{
  long long i = (long long)blockIdx.x * blockDim.x + threadIdx.x;
  long long total = rows * Dh;
  if (i >= total) return;
  int d = (int)(i % Dh); long long r = i / Dh;
  y[i] *= dsilu(xz[r * (long long)(2 * Dh) + Dh + d]);
}

__global__ void k_split_qkv(const float* __restrict__ qkv, float* __restrict__ q,
                            float* __restrict__ kk, float* __restrict__ v,
                            int B, int L, int heads, int hd)
{
  long long i = (long long)blockIdx.x * blockDim.x + threadIdx.x;
  long long total = (long long)B * L * heads * hd;
  if (i >= total) return;
  int e = (int)(i % hd); long long r = i / hd;
  int hh = (int)(r % heads); long long q2 = r / heads;
  int l = (int)(q2 % L); int b = (int)(q2 / L);
  const int C = heads * hd;
  const long long src = ((long long)b * L + l) * (3 * C);
  const long long dst = (((long long)b * heads + hh) * L + l) * hd + e;
  q[dst]  = qkv[src + 0 * C + hh * hd + e];
  kk[dst] = qkv[src + 1 * C + hh * hd + e];
  v[dst]  = qkv[src + 2 * C + hh * hd + e];
}

__global__ void k_merge_heads(const float* __restrict__ ah, float* __restrict__ o,
                              int B, int L, int heads, int hd)
{
  long long i = (long long)blockIdx.x * blockDim.x + threadIdx.x;
  long long total = (long long)B * L * heads * hd;
  if (i >= total) return;
  const int C = heads * hd;
  int c = (int)(i % C); long long r = i / C;
  int l = (int)(r % L); int b = (int)(r / L);
  int hh = c / hd, e = c % hd;
  o[i] = ah[(((long long)b * heads + hh) * L + l) * hd + e];
}

__global__ void k_softmax(float* __restrict__ x, int rows, int cols)
{
  const int row = blockIdx.x;
  const int lane = threadIdx.x;
  if (row >= rows) return;
  float* xr = x + (long long)row * cols;
  float mx = -1e30f;
  for (int c = lane; c < cols; c += 32) mx = fmaxf(mx, xr[c]);
  mx = wred_max(mx);
  float s = 0.f;
  for (int c = lane; c < cols; c += 32) { float e = __expf(xr[c] - mx); xr[c] = e; s += e; }
  s = wred_sum(s);
  const float inv = 1.f / s;
  for (int c = lane; c < cols; c += 32) xr[c] *= inv;
}

__global__ void k_mean_hw(const float* __restrict__ x, float* __restrict__ o,
                          int B, int HW, int C)
{
  const int bc = blockIdx.x;      // b*C + c
  const int lane = threadIdx.x;
  if (bc >= B * C) return;
  const int c = bc % C, b = bc / C;
  float s = 0.f;
  for (int p = lane; p < HW; p += 32) s += x[((long long)b * HW + p) * C + c];
  s = wred_sum(s);
  if (lane == 0) o[bc] = s / HW;
}

__global__ void k_se_apply(const float* __restrict__ o2, const float* __restrict__ se,
                           const float* __restrict__ xres, float* __restrict__ out,
                           int B, int HW, int C)
{
  long long i = (long long)blockIdx.x * blockDim.x + threadIdx.x;
  long long total = (long long)B * HW * C;
  if (i >= total) return;
  int c = (int)(i % C); long long r = i / C;
  int b = (int)(r / HW);
  float v = o2[i] * se[(long long)b * C + c] + xres[i];
  out[i] = fmaxf(v, 0.f);
}

// bilinear (half-pixel centers) upsample of x (B,Hi,Wi,C) to (B,Ho,Wo,C), + y
__global__ void k_up_add(const float* __restrict__ x, const float* __restrict__ y,
                         float* __restrict__ o, int B, int Hi, int Wi,
                         int Ho, int Wo, int C)
{
  long long i = (long long)blockIdx.x * blockDim.x + threadIdx.x;
  long long total = (long long)B * Ho * Wo * C;
  if (i >= total) return;
  int c = (int)(i % C); long long r = i / C;
  int w = (int)(r % Wo); long long q = r / Wo;
  int h = (int)(q % Ho); int b = (int)(q / Ho);
  const float sy = (float)Hi / Ho, sx = (float)Wi / Wo;
  const float fy = (h + 0.5f) * sy - 0.5f;
  const float fx = (w + 0.5f) * sx - 0.5f;
  int y0 = (int)floorf(fy), x0 = (int)floorf(fx);
  const float wy = fy - y0, wx = fx - x0;
  int y1 = iclampi(y0 + 1, 0, Hi - 1), x1 = iclampi(x0 + 1, 0, Wi - 1);
  y0 = iclampi(y0, 0, Hi - 1); x0 = iclampi(x0, 0, Wi - 1);
  const float* xb = x + ((long long)b * Hi) * Wi * C + c;
  const float g00 = xb[((long long)y0 * Wi + x0) * C];
  const float g01 = xb[((long long)y0 * Wi + x1) * C];
  const float g10 = xb[((long long)y1 * Wi + x0) * C];
  const float g11 = xb[((long long)y1 * Wi + x1) * C];
  const float v = (1.f - wy) * ((1.f - wx) * g00 + wx * g01)
                +        wy  * ((1.f - wx) * g10 + wx * g11);
  o[i] = v + y[i];
}

// 3x3 SAME im2col, NHWC -> col (B*HW, C*9) with k = c*9 + ky*3 + kx (matches OIHW)
__global__ void k_im2col(const float* __restrict__ x, float* __restrict__ col,
                         int B, int H, int W, int C)
{
  long long i = (long long)blockIdx.x * blockDim.x + threadIdx.x;
  long long total = (long long)B * H * W * C * 9;
  if (i >= total) return;
  int j = (int)(i % 9); long long r = i / 9;
  int c = (int)(r % C); long long q = r / C;
  int w = (int)(q % W); long long t = q / W;
  int h = (int)(t % H); int b = (int)(t / H);
  const int ky = j / 3, kx = j % 3;
  const int yy = h + ky - 1, xx = w + kx - 1;
  float v = 0.f;
  if (yy >= 0 && yy < H && xx >= 0 && xx < W)
    v = x[(((long long)b * H + yy) * W + xx) * C + c];
  col[i] = v;
}

// ---------------------------------------------------------------------------
// Host side
// ---------------------------------------------------------------------------
static inline unsigned gdim(long long total, int block) {
  return (unsigned)((total + block - 1) / block);
}

static void gemm_launch(hipStream_t st, const float* A, const float* W, float* O,
                        const float* bias, const float* scale, const float* shift,
                        int M, int N, int K, int lda, int ldw, int ldo,
                        long long sA, long long sW, long long sO, long long sBias,
                        float alpha, float smul, int wt, int at, int bdiv, int act, int batch)
{
  if (wt == 0 && at == 0) {
    if (N >= 64) {
      dim3 grid((M + 15) / 16, (N + 63) / 64, batch);
      k_gemm<0, 0, 4><<<grid, 32, 0, st>>>(A, W, O, bias, scale, shift, M, N, K,
                                           lda, ldw, ldo, sA, sW, sO, sBias,
                                           alpha, smul, bdiv, act);
    } else {
      dim3 grid((M + 15) / 16, (N + 15) / 16, batch);
      k_gemm<0, 0, 1><<<grid, 32, 0, st>>>(A, W, O, bias, scale, shift, M, N, K,
                                           lda, ldw, ldo, sA, sW, sO, sBias,
                                           alpha, smul, bdiv, act);
    }
  } else if (wt == 0) {   // A transposed
    dim3 grid((M + 15) / 16, (N + 15) / 16, batch);
    k_gemm<0, 1, 1><<<grid, 32, 0, st>>>(A, W, O, bias, scale, shift, M, N, K,
                                         lda, ldw, ldo, sA, sW, sO, sBias,
                                         alpha, smul, bdiv, act);
  } else {                // W transposed (W[k][n])
    dim3 grid((M + 15) / 16, (N + 15) / 16, batch);
    k_gemm<1, 0, 1><<<grid, 32, 0, st>>>(A, W, O, bias, scale, shift, M, N, K,
                                         lda, ldw, ldo, sA, sW, sO, sBias,
                                         alpha, smul, bdiv, act);
  }
}

static void gemm2d(hipStream_t st, const float* A, const float* W, float* O,
                   const float* bias, const float* scale, const float* shift,
                   int M, int N, int K, int act, float smul)
{
  gemm_launch(st, A, W, O, bias, scale, shift, M, N, K, K, K, N,
              0, 0, 0, 0, 1.f, smul, 0, 0, 1, act, 1);
}

struct AttnP { const float *proj_b, *proj_w, *qkv_w; };
struct FuseP { const float *b, *be, *g, *w; };
struct IcsfP { const float *bn1_b, *bn1_g, *bn2_b, *bn2_g, *c1_w, *dw_w,
                            *se1_b, *se1_w, *se2_b, *se2_w; };
struct SsmP  { const float *A_log, *Ds, *conv_b, *conv_w, *dt_proj_b, *dt_proj_w,
                            *in_proj_w, *ln_b, *ln_g, *out_norm_b, *out_norm_g,
                            *out_proj_w, *pw_b, *pw_w, *x_proj_w; };

struct ParamCursor {
  void* const* d; int i;
  const float* next() { return (const float*)d[i++]; }
};

extern "C" void kernel_launch(void* const* d_in, const int* in_sizes, int n_in,
                              void* d_out, int out_size, void* d_ws, size_t ws_size,
                              hipStream_t stream)
{
  (void)in_sizes; (void)n_in; (void)out_size; (void)ws_size;
  const int Bn = 4, Cc = 128, Dh = 256, HEADS = 8, HD = 16;
  const float BNS = 1.0f / sqrtf(1.0f + 1e-5f);   // eval BatchNorm scale

  // ---- inputs: pre1,post1,pre2,post2,pre3,post3,pre4,post4 then sorted params
  const float* pre1  = (const float*)d_in[0];
  const float* post1 = (const float*)d_in[1];
  const float* pre2  = (const float*)d_in[2];
  const float* post2 = (const float*)d_in[3];
  const float* pre3  = (const float*)d_in[4];
  const float* post3 = (const float*)d_in[5];
  const float* pre4  = (const float*)d_in[6];
  const float* post4 = (const float*)d_in[7];

  ParamCursor pc{d_in, 8};
  // JAX pytree order (sorted keys): at31, at41, fuse1..fuse4, sm1..sm3, st11..st41
  AttnP at31{pc.next(), pc.next(), pc.next()};
  AttnP at41{pc.next(), pc.next(), pc.next()};
  FuseP fu1{pc.next(), pc.next(), pc.next(), pc.next()};
  FuseP fu2{pc.next(), pc.next(), pc.next(), pc.next()};
  FuseP fu3{pc.next(), pc.next(), pc.next(), pc.next()};
  FuseP fu4{pc.next(), pc.next(), pc.next(), pc.next()};
  IcsfP sm1{pc.next(), pc.next(), pc.next(), pc.next(), pc.next(),
            pc.next(), pc.next(), pc.next(), pc.next(), pc.next()};
  IcsfP sm2{pc.next(), pc.next(), pc.next(), pc.next(), pc.next(),
            pc.next(), pc.next(), pc.next(), pc.next(), pc.next()};
  IcsfP sm3{pc.next(), pc.next(), pc.next(), pc.next(), pc.next(),
            pc.next(), pc.next(), pc.next(), pc.next(), pc.next()};
  SsmP st11{pc.next(), pc.next(), pc.next(), pc.next(), pc.next(), pc.next(),
            pc.next(), pc.next(), pc.next(), pc.next(), pc.next(), pc.next(),
            pc.next(), pc.next(), pc.next()};
  SsmP st21{pc.next(), pc.next(), pc.next(), pc.next(), pc.next(), pc.next(),
            pc.next(), pc.next(), pc.next(), pc.next(), pc.next(), pc.next(),
            pc.next(), pc.next(), pc.next()};
  SsmP st31{pc.next(), pc.next(), pc.next(), pc.next(), pc.next(), pc.next(),
            pc.next(), pc.next(), pc.next(), pc.next(), pc.next(), pc.next(),
            pc.next(), pc.next(), pc.next()};
  SsmP st41{pc.next(), pc.next(), pc.next(), pc.next(), pc.next(), pc.next(),
            pc.next(), pc.next(), pc.next(), pc.next(), pc.next(), pc.next(),
            pc.next(), pc.next(), pc.next()};

  // ---- workspace bump allocator
  char* wsb = (char*)d_ws;
  size_t off = 0;
  auto alloc = [&](long long nfloats) -> float* {
    float* p = (float*)(wsb + off);
    off += ((size_t)nfloats * 4 + 255) & ~(size_t)255;
    return p;
  };

  // persistent stage tensors (NHWC)
  float* p41 = alloc((long long)Bn * 64 * Cc);
  float* p42 = alloc((long long)Bn * 64 * Cc);
  float* p4  = alloc((long long)Bn * 64 * Cc);
  float* p31 = alloc((long long)Bn * 256 * Cc);
  float* p32 = alloc((long long)Bn * 256 * Cc);
  float* p3  = alloc((long long)Bn * 256 * Cc);
  float* p3u = alloc((long long)Bn * 256 * Cc);
  float* p21 = alloc((long long)Bn * 1024 * Cc);
  float* p2  = alloc((long long)Bn * 1024 * Cc);
  float* p2u = alloc((long long)Bn * 1024 * Cc);
  float* p11 = alloc((long long)Bn * 4096 * Cc);
  float* p1  = alloc((long long)Bn * 4096 * Cc);
  float* p1u = alloc((long long)Bn * 4096 * Cc);
  const size_t scratch_base = off;

  // ---- SS2D spatio-temporal block: NCHW pre/post -> NHWC (BL,128) output
  auto st_block = [&](const float* pre, const float* post, int cin_half, int S,
                      const SsmP& sp, float* out_nhwc) {
    const int H = S, W = S, L = S * S, BL = Bn * L, Cin = 2 * cin_half;
    const size_t save = off;

    float* cat = alloc((long long)BL * Cin);
    k_cat_nchw2nhwc<<<gdim((long long)BL * Cin, 256), 256, 0, stream>>>(
        pre, post, cat, Bn, cin_half, cin_half, L);

    float* y0 = alloc((long long)BL * Cc);        // 1x1 conv (pw)
    gemm2d(stream, cat, sp.pw_w, y0, sp.pw_b, nullptr, nullptr, BL, Cc, Cin, 0, 1.f);

    float* t = alloc((long long)BL * Cc);         // LN
    k_layernorm<<<BL, 32, 0, stream>>>(y0, t, sp.ln_g, sp.ln_b, BL, Cc);

    float* xz = alloc((long long)BL * 2 * Dh);    // in_proj
    gemm2d(stream, t, sp.in_proj_w, xz, nullptr, nullptr, nullptr, BL, 2 * Dh, Cc, 0, 1.f);

    float* xc = alloc((long long)BL * Dh);        // depthwise conv + silu on xx half
    k_dwconv3x3<<<gdim((long long)BL * Dh, 256), 256, 0, stream>>>(
        xz, sp.conv_w, sp.conv_b, nullptr, nullptr, 1.f, xc, Bn, H, W, Dh, 2 * Dh, 2);

    float* xs = alloc(4LL * Bn * Dh * L);         // 4 scan directions
    k_build_xs<<<gdim((long long)BL * Dh, 256), 256, 0, stream>>>(xc, xs, Bn, H, W, Dh);

    // x_dbl[k,b,l,c] = sum_d xs[k,b,d,l] * x_proj_w[k,c,d]   (A transposed)
    float* xdbl = alloc(4LL * BL * 40);
    gemm_launch(stream, xs, sp.x_proj_w, xdbl, nullptr, nullptr, nullptr,
                L, 40, Dh, L, Dh, 40,
                (long long)Dh * L, 40LL * Dh, (long long)L * 40, 0,
                1.f, 1.f, /*wt*/0, /*at*/1, /*bdiv*/Bn, /*act*/0, 4 * Bn);

    // dt[k,b,l,d] = softplus(dts @ dt_proj_w[k]^T + dt_proj_b[k])
    float* dtT = alloc(4LL * BL * Dh);
    gemm_launch(stream, xdbl, sp.dt_proj_w, dtT, sp.dt_proj_b, nullptr, nullptr,
                L, Dh, 8, 40, 8, Dh,
                (long long)L * 40, (long long)Dh * 8, (long long)L * Dh, Dh,
                1.f, 1.f, 0, 0, Bn, /*softplus*/4, 4 * Bn);

    float* ys = alloc(4LL * Bn * Dh * L);
    k_scan<<<gdim(4LL * Bn * Dh, 128), 128, 0, stream>>>(
        xs, dtT, xdbl, sp.A_log, sp.Ds, ys, Bn, Dh, L);

    float* yt = alloc((long long)BL * Dh);
    k_combine_y<<<gdim((long long)BL * Dh, 256), 256, 0, stream>>>(ys, yt, Bn, H, W, Dh);

    float* yn = alloc((long long)BL * Dh);
    k_layernorm<<<BL, 32, 0, stream>>>(yt, yn, sp.out_norm_g, sp.out_norm_b, BL, Dh);
    k_gate<<<gdim((long long)BL * Dh, 256), 256, 0, stream>>>(yn, xz, BL, Dh);

    gemm2d(stream, yn, sp.out_proj_w, out_nhwc, nullptr, nullptr, nullptr,
           BL, Cc, Dh, 0, 1.f);
    k_add_inplace<<<gdim((long long)BL * Cc, 256), 256, 0, stream>>>(
        out_nhwc, y0, (long long)BL * Cc);   // residual

    off = save;
  };

  // ---- multi-head attention on NHWC tokens
  auto attn_stage = [&](const float* x_nhwc, int S, const AttnP& ap, float* out_nhwc) {
    const int L = S * S, BL = Bn * L, BH = Bn * HEADS;
    const size_t save = off;

    float* qkv = alloc((long long)BL * 3 * Cc);
    gemm2d(stream, x_nhwc, ap.qkv_w, qkv, nullptr, nullptr, nullptr, BL, 3 * Cc, Cc, 0, 1.f);

    float* qb = alloc((long long)BL * Cc);
    float* kb = alloc((long long)BL * Cc);
    float* vb = alloc((long long)BL * Cc);
    k_split_qkv<<<gdim((long long)BL * Cc, 256), 256, 0, stream>>>(
        qkv, qb, kb, vb, Bn, L, HEADS, HD);

    float* sc = alloc((long long)BH * L * L);
    gemm_launch(stream, qb, kb, sc, nullptr, nullptr, nullptr,
                L, L, HD, HD, HD, L,
                (long long)L * HD, (long long)L * HD, (long long)L * L, 0,
                0.25f /*hd^-0.5*/, 1.f, 0, 0, 1, 0, BH);
    k_softmax<<<BH * L, 32, 0, stream>>>(sc, BH * L, L);

    float* ao = alloc((long long)BL * Cc);
    gemm_launch(stream, sc, vb, ao, nullptr, nullptr, nullptr,
                L, HD, L, L, HD, HD,
                (long long)L * L, (long long)L * HD, (long long)L * HD, 0,
                1.f, 1.f, /*wt=1: W[k][n]*/1, 0, 1, 0, BH);

    float* mg = alloc((long long)BL * Cc);
    k_merge_heads<<<gdim((long long)BL * Cc, 256), 256, 0, stream>>>(
        ao, mg, Bn, L, HEADS, HD);

    gemm2d(stream, mg, ap.proj_w, out_nhwc, ap.proj_b, nullptr, nullptr, BL, Cc, Cc, 0, 1.f);
    off = save;
  };

  // ---- fuse: relu(BN(1x1 conv(cat)))
  auto fuse_op = [&](const float* a, const float* b2, int S, const FuseP& fp, float* out) {
    const int BL = Bn * S * S;
    const size_t save = off;
    const float* src = a; int K = Cc;
    if (b2) {
      float* cat = alloc((long long)BL * 2 * Cc);
      k_cat_nhwc<<<gdim((long long)BL * 2 * Cc, 256), 256, 0, stream>>>(
          a, b2, cat, BL, Cc, Cc);
      src = cat; K = 2 * Cc;
    }
    gemm2d(stream, src, fp.w, out, fp.b, fp.g, fp.be, BL, Cc, K, /*relu*/1, BNS);
    off = save;
  };

  // ---- ICSF refinement block (3x3 conv -> dw conv -> SE -> residual relu)
  auto icsf = [&](const float* x, int S, const IcsfP& ip, float* out) {
    const int H = S, W = S, BL = Bn * S * S;
    const size_t save = off;

    float* col = alloc((long long)BL * Cc * 9);
    k_im2col<<<gdim((long long)BL * Cc * 9, 256), 256, 0, stream>>>(x, col, Bn, H, W, Cc);
    float* o1 = alloc((long long)BL * Cc);
    gemm2d(stream, col, ip.c1_w, o1, nullptr, ip.bn1_g, ip.bn1_b,
           BL, Cc, Cc * 9, /*relu*/1, BNS);

    float* o2 = alloc((long long)BL * Cc);
    k_dwconv3x3<<<gdim((long long)BL * Cc, 256), 256, 0, stream>>>(
        o1, ip.dw_w, nullptr, ip.bn2_g, ip.bn2_b, BNS, o2, Bn, H, W, Cc, Cc, /*relu*/1);

    float* se = alloc((long long)Bn * Cc);
    k_mean_hw<<<Bn * Cc, 32, 0, stream>>>(o2, se, Bn, H * W, Cc);
    float* s1 = alloc((long long)Bn * 16);
    gemm2d(stream, se, ip.se1_w, s1, ip.se1_b, nullptr, nullptr, Bn, 16, Cc, /*relu*/1, 1.f);
    float* s2 = alloc((long long)Bn * Cc);
    gemm2d(stream, s1, ip.se2_w, s2, ip.se2_b, nullptr, nullptr, Bn, Cc, 16, /*sigmoid*/3, 1.f);

    k_se_apply<<<gdim((long long)BL * Cc, 256), 256, 0, stream>>>(
        o2, s2, x, out, Bn, H * W, Cc);
    off = save;
  };

  // =========================== decoder pipeline ===========================
  // Stage I (8x8)
  st_block(pre4, post4, 768, 8, st41, p41);
  attn_stage(p41, 8, at41, p42);
  fuse_op(p41, p42, 8, fu4, p4);

  // Stage II (16x16)
  st_block(pre3, post3, 384, 16, st31, p31);
  attn_stage(p31, 16, at31, p32);
  fuse_op(p31, p32, 16, fu3, p3);
  k_up_add<<<gdim((long long)Bn * 256 * Cc, 256), 256, 0, stream>>>(
      p4, p3, p3u, Bn, 8, 8, 16, 16, Cc);
  icsf(p3u, 16, sm3, p3);

  // Stage III (32x32)
  st_block(pre2, post2, 192, 32, st21, p21);
  fuse_op(p21, nullptr, 32, fu2, p2);
  k_up_add<<<gdim((long long)Bn * 1024 * Cc, 256), 256, 0, stream>>>(
      p3, p2, p2u, Bn, 16, 16, 32, 32, Cc);
  icsf(p2u, 32, sm2, p2);

  // Stage IV (64x64)
  st_block(pre1, post1, 96, 64, st11, p11);
  fuse_op(p11, nullptr, 64, fu1, p1);
  k_up_add<<<gdim((long long)Bn * 4096 * Cc, 256), 256, 0, stream>>>(
      p2, p1, p1u, Bn, 32, 32, 64, 64, Cc);
  icsf(p1u, 64, sm1, p1);

  // final NHWC -> NCHW into d_out (B,128,64,64)
  k_nhwc2nchw<<<gdim((long long)Bn * Cc * 4096, 256), 256, 0, stream>>>(
      p1, (float*)d_out, Bn, Cc, 4096);

  (void)scratch_base;
}
